// RGNN_Model_850403525019
// MI455X (gfx1250) — compile-verified, matrix operand order checked by reading, not run on hardware
//
#include <hip/hip_runtime.h>

// ---------------------------------------------------------------------------
// Types for CDNA5 WMMA (wave32): bf16 16x16x32 -> f32 16x16 accumulator
// ---------------------------------------------------------------------------
typedef __attribute__((ext_vector_type(16))) __bf16 v16bf;
typedef __attribute__((ext_vector_type(8)))  float  v8f;

union Frag {
    v16bf v;
    uint4 q[2];   // 2 x 16B = 8 VGPRs = 16 bf16 per lane
};

// Load a 16x32 bf16 operand fragment from a row-major [rows, ld] buffer.
// Works for A (row = M) and for B when the source is stored transposed
// (row = N, i.e. B^T[N,K] row-major). Matches ISA 16-bit operand layout:
//   lanes 0-15 : K = k0+0..7   (q[0]) and K = k0+16..23 (q[1])
//   lanes 16-31: K = k0+8..15  (q[0]) and K = k0+24..31 (q[1])
template <typename PT>
__device__ __forceinline__ void loadFragRM(Frag& f, PT base, int ld,
                                           int row0, int k0, int lane) {
    int r  = row0 + (lane & 15);
    int kb = k0 + ((lane >> 4) << 3);
    const __bf16* p = base + r * ld + kb;
    f.q[0] = *(const uint4*)(p);
    f.q[1] = *(const uint4*)(p + 16);
}

__device__ __forceinline__ v8f wmma_bf16(const Frag& a, const Frag& b, v8f c) {
    return __builtin_amdgcn_wmma_f32_16x16x32_bf16(
        /*neg_a=*/false, a.v, /*neg_b=*/false, b.v,
        /*c_mod=*/(short)0, c, /*reuse_a=*/false, /*reuse_b=*/false);
}

// ---------------------------------------------------------------------------
// Kernel A: build normalized adjacency An = D^-1/2 (W_sym + I) D^-1/2 -> bf16
// ---------------------------------------------------------------------------
__global__ __launch_bounds__(256) void build_an(const float* __restrict__ tril,
                                                __bf16* __restrict__ An) {
    __shared__ float Wm[64][65];
    __shared__ float dinv[64];
    const int tid = threadIdx.x;
    for (int i = tid; i < 64 * 64; i += 256) {
        int r = i >> 6, c = i & 63;
        int hi = r >= c ? r : c;
        int lo = r >= c ? c : r;
        Wm[r][c] = tril[hi * (hi + 1) / 2 + lo];   // symmetric reconstruction
    }
    __syncthreads();
    if (tid < 64) {
        float d = 1.0f;                            // |self-loop weight| = 1
        for (int c = 0; c < 64; ++c) d += fabsf(Wm[tid][c]);
        dinv[tid] = (d > 0.0f) ? rsqrtf(d) : 0.0f;
    }
    __syncthreads();
    for (int i = tid; i < 64 * 64; i += 256) {
        int r = i >> 6, c = i & 63;
        float a = Wm[r][c] + (r == c ? 1.0f : 0.0f);
        An[i] = (__bf16)(dinv[r] * a * dinv[c]);   // row-major [64][64]
    }
}

// ---------------------------------------------------------------------------
// Kernel W: conv_W [256,512] f32 -> WT [512,256] bf16 (B^T layout for WMMA)
// ---------------------------------------------------------------------------
__global__ __launch_bounds__(256) void transpose_w(const float* __restrict__ Wc,
                                                   __bf16* __restrict__ WT) {
    int i = blockIdx.x * 256 + threadIdx.x;
    if (i < 256 * 512) {
        int f = i >> 9;          // 0..255
        int h = i & 511;         // 0..511
        WT[h * 256 + f] = (__bf16)Wc[i];
    }
}

// ---------------------------------------------------------------------------
// Kernel B: fully fused per-batch pipeline (1 block = 1 graph, 8 waves)
//   Xb --(bf16, transposed)--> LDS
//   Y1 = An @ Xb   (store Y1^T in LDS)
//   Y2 = An @ Y1   (store row-major in LDS)
//   H  = Y2 @ convW ; relu(H + b) ; pooled[b] = sum over 64 nodes
// ---------------------------------------------------------------------------
__global__ __launch_bounds__(256) void sgconv_fused(
    const float*  __restrict__ X,       // [2048,64,256] f32
    const __bf16* __restrict__ An,      // [64,64] bf16 row-major
    const __bf16* __restrict__ WT,      // [512,256] bf16 (conv_W^T)
    const float*  __restrict__ conv_b,  // [512]
    float*        __restrict__ pooled)  // [2048,512]
{
    // LDS: An [64][72] ; buf0 = Xt[256][72] later reused as Y2[64][264] ;
    //      buf1 = Y1t[256][72].  Total ~81 KB.
    __shared__ __bf16 smem[64 * 72 + 256 * 72 + 256 * 72];
    __bf16* sAn  = smem;
    __bf16* buf0 = smem + 64 * 72;
    __bf16* buf1 = buf0 + 256 * 72;

    const int tid  = threadIdx.x;
    const int lane = tid & 31;
    const int wave = tid >> 5;       // 0..7
    const int b    = blockIdx.x;

    // stage An -> LDS (padded ld=72 to spread banks, rows stay 16B aligned)
    for (int i = tid; i < 64 * 64; i += 256)
        sAn[(i >> 6) * 72 + (i & 63)] = An[i];

    // stage Xb, convert f32->bf16, store TRANSPOSED (feature-major) for B^T use
    const float4* Xb = (const float4*)(X + (size_t)b * 64 * 256);
    for (int i = tid; i < 64 * 64; i += 256) {       // 4096 float4 loads
        int node = i >> 6;
        int f4   = i & 63;
        float4 vv = Xb[node * 64 + f4];
        int f = f4 * 4;
        buf0[(f + 0) * 72 + node] = (__bf16)vv.x;
        buf0[(f + 1) * 72 + node] = (__bf16)vv.y;
        buf0[(f + 2) * 72 + node] = (__bf16)vv.z;
        buf0[(f + 3) * 72 + node] = (__bf16)vv.w;
    }
    __syncthreads();

    // All 8 An A-fragments (mt 0..3, k {0,32}) are reused by both hops and
    // every ntl iteration: load once, keep resident (~64 VGPRs).
    Frag aF[8];
    #pragma unroll
    for (int mt = 0; mt < 4; ++mt) {
        loadFragRM(aF[2 * mt + 0], (const __bf16*)sAn, 72, mt * 16, 0,  lane);
        loadFragRM(aF[2 * mt + 1], (const __bf16*)sAn, 72, mt * 16, 32, lane);
    }

    // ---- hop 1: Y1 = An @ X  (M=64, N=256, K=64) -> Y1^T in buf1 ----
    for (int ntl = 0; ntl < 2; ++ntl) {
        const int nt = wave * 2 + ntl;               // feature tile 0..15
        Frag bf0, bf1;
        loadFragRM(bf0, (const __bf16*)buf0, 72, nt * 16, 0,  lane);
        loadFragRM(bf1, (const __bf16*)buf0, 72, nt * 16, 32, lane);
        #pragma unroll
        for (int mt = 0; mt < 4; ++mt) {
            v8f acc = {};
            acc = wmma_bf16(aF[2 * mt + 0], bf0, acc);
            acc = wmma_bf16(aF[2 * mt + 1], bf1, acc);
            int n  = nt * 16 + (lane & 15);
            int mb = mt * 16 + 8 * (lane >> 4);
            #pragma unroll
            for (int r = 0; r < 8; ++r)
                buf1[n * 72 + mb + r] = (__bf16)acc[r];   // transposed store
        }
    }
    __syncthreads();

    // ---- hop 2: Y2 = An @ Y1 -> row-major [64][264] in buf0 ----
    for (int ntl = 0; ntl < 2; ++ntl) {
        const int nt = wave * 2 + ntl;
        Frag bf0, bf1;
        loadFragRM(bf0, (const __bf16*)buf1, 72, nt * 16, 0,  lane);
        loadFragRM(bf1, (const __bf16*)buf1, 72, nt * 16, 32, lane);
        #pragma unroll
        for (int mt = 0; mt < 4; ++mt) {
            v8f acc = {};
            acc = wmma_bf16(aF[2 * mt + 0], bf0, acc);
            acc = wmma_bf16(aF[2 * mt + 1], bf1, acc);
            int nb = nt * 16 + (lane & 15);
            int mb = mt * 16 + 8 * (lane >> 4);
            #pragma unroll
            for (int r = 0; r < 8; ++r)
                buf0[(mb + r) * 264 + nb] = (__bf16)acc[r];  // row-major store
        }
    }
    __syncthreads();

    // ---- conv + ReLU + pool: H = Y2 @ convW (M=64, N=512, K=256) ----
    // j -> kt -> mt with 4 persistent accumulators: each WT B-fragment is
    // loaded ONCE per (j,kt) and feeds 4 independent WMMAs (hazard hiding,
    // 4x fewer global loads than reloading inside the mt loop).
    for (int j = 0; j < 4; ++j) {
        const int nt = wave * 4 + j;                 // output-feature tile 0..31
        const float bias = conv_b[nt * 16 + (lane & 15)];
        v8f acc0 = {}, acc1 = {}, acc2 = {}, acc3 = {};
        #pragma unroll
        for (int kt = 0; kt < 8; ++kt) {
            Frag bb;
            loadFragRM(bb, WT, 256, nt * 16, kt * 32, lane);
            Frag a0, a1, a2, a3;
            loadFragRM(a0, (const __bf16*)buf0, 264, 0,  kt * 32, lane);
            loadFragRM(a1, (const __bf16*)buf0, 264, 16, kt * 32, lane);
            loadFragRM(a2, (const __bf16*)buf0, 264, 32, kt * 32, lane);
            loadFragRM(a3, (const __bf16*)buf0, 264, 48, kt * 32, lane);
            acc0 = wmma_bf16(a0, bb, acc0);
            acc1 = wmma_bf16(a1, bb, acc1);
            acc2 = wmma_bf16(a2, bb, acc2);
            acc3 = wmma_bf16(a3, bb, acc3);
        }
        float s = 0.0f;
        #pragma unroll
        for (int r = 0; r < 8; ++r) {                // relu + node-sum
            float v0 = acc0[r] + bias; s += v0 > 0.0f ? v0 : 0.0f;
            float v1 = acc1[r] + bias; s += v1 > 0.0f ? v1 : 0.0f;
            float v2 = acc2[r] + bias; s += v2 > 0.0f ? v2 : 0.0f;
            float v3 = acc3[r] + bias; s += v3 > 0.0f ? v3 : 0.0f;
        }
        s += __shfl_down(s, 16, 32);                 // fold the two half-waves
        if (lane < 16)
            pooled[(size_t)b * 512 + nt * 16 + lane] = s;
    }
}

// ---------------------------------------------------------------------------
// Kernel C: logits = pooled @ fc_W^T + fc_b ; log_softmax (C=2)
// one wave per batch row
// ---------------------------------------------------------------------------
__global__ __launch_bounds__(256) void fc_logsoftmax(
    const float* __restrict__ pooled, const float* __restrict__ fcW,
    const float* __restrict__ fcb, float* __restrict__ out) {
    const int wave = threadIdx.x >> 5;
    const int lane = threadIdx.x & 31;
    const int b = blockIdx.x * 8 + wave;
    const float* p = pooled + (size_t)b * 512;
    float s0 = 0.0f, s1 = 0.0f;
    for (int k = lane; k < 512; k += 32) {
        float x = p[k];
        s0 += x * fcW[k];
        s1 += x * fcW[512 + k];
    }
    #pragma unroll
    for (int off = 16; off; off >>= 1) {
        s0 += __shfl_down(s0, off, 32);
        s1 += __shfl_down(s1, off, 32);
    }
    if (lane == 0) {
        float l0 = s0 + fcb[0];
        float l1 = s1 + fcb[1];
        float m  = fmaxf(l0, l1);
        float lse = m + logf(expf(l0 - m) + expf(l1 - m));
        out[b * 2 + 0] = l0 - lse;
        out[b * 2 + 1] = l1 - lse;
    }
}

// ---------------------------------------------------------------------------
extern "C" void kernel_launch(void* const* d_in, const int* in_sizes, int n_in,
                              void* d_out, int out_size, void* d_ws, size_t ws_size,
                              hipStream_t stream) {
    const float* X     = (const float*)d_in[0];   // [2048,64,256]
    const float* tril  = (const float*)d_in[1];   // [2080]
    const float* convW = (const float*)d_in[2];   // [256,512]
    const float* convb = (const float*)d_in[3];   // [512]
    const float* fcW   = (const float*)d_in[4];   // [2,512]
    const float* fcb   = (const float*)d_in[5];   // [2]

    char* ws = (char*)d_ws;
    __bf16* An = (__bf16*)ws;                              //   8 KB
    __bf16* WT = (__bf16*)(ws + 8192);                     // 256 KB
    float* pooled = (float*)(ws + 8192 + 262144);          //   4 MB

    build_an<<<1, 256, 0, stream>>>(tril, An);
    transpose_w<<<512, 256, 0, stream>>>(convW, WT);
    sgconv_fused<<<2048, 256, 0, stream>>>(X, An, WT, convb, pooled);
    fc_logsoftmax<<<256, 256, 0, stream>>>(pooled, fcW, fcb, (float*)d_out);
}
